// PedusGPT_34411277976015
// MI455X (gfx1250) — matrix operand order repaired and checked by reading, not compiled, not run on hardware
//
#include <hip/hip_runtime.h>
#include <hip/hip_bf16.h>
#include <math.h>

// GPT-small forward for MI455X (gfx1250, wave32, WMMA bf16 + TDM staging).
#define Dm    768
#define Tm    2048
#define Bmb   2
#define Hm    12
#define Lm    6
#define Vm    32000
#define Mrows 4096   // B*T

typedef __attribute__((ext_vector_type(16))) __bf16 v16bf;
typedef __attribute__((ext_vector_type(8)))  float  v8f;
typedef __attribute__((ext_vector_type(4)))  unsigned int v4u;
typedef __attribute__((ext_vector_type(8)))  int v8i;
typedef __attribute__((ext_vector_type(4)))  int v4i;

#if defined(__has_builtin)
#  if __has_builtin(__builtin_amdgcn_tensor_load_to_lds) && \
      __has_builtin(__builtin_amdgcn_s_wait_tensorcnt)
#    define PEDUS_HAVE_TDM 1
#  endif
#endif
#ifndef PEDUS_HAVE_TDM
#  define PEDUS_HAVE_TDM 0
#endif

struct alignas(16) B16x8 { unsigned long long lo, hi; };
union Frag16 { v16bf v; B16x8 h[2]; };

// A-fragment (16x32 bf16): lane m = l%16, h = l/16; K = {8h..8h+7} U {8h+16..8h+23}
static __device__ inline v16bf load_a_frag(const __bf16* base, int stride, int row, int lane) {
  int h = (lane >> 4) & 1;
  const __bf16* p = base + row * stride + 8 * h;
  Frag16 f;
  f.h[0] = *(const B16x8*)(p);
  f.h[1] = *(const B16x8*)(p + 16);
  return f.v;
}
// B-fragment (32x16 bf16, staged as [N][K]): lane n = l%16, h = l/16; K = {8h..8h+7} U {8h+16..8h+23}
static __device__ inline v16bf load_b_frag(const __bf16* base, int stride, int row, int lane) {
  int h = (lane >> 4) & 1;
  const __bf16* p = base + row * stride + 8 * h;
  Frag16 f;
  f.h[0] = *(const B16x8*)(p);
  f.h[1] = *(const B16x8*)(p + 16);
  return f.v;
}
#define WMMA_BF16(a, b, c) \
  __builtin_amdgcn_wmma_f32_16x16x32_bf16(false, (a), false, (b), (short)0, (c), false, false)

#if PEDUS_HAVE_TDM
// Tensor Data Mover: DMA a 2D bf16 tile [tile_rows][tile_k] from global (row
// stride = row_stride elements) into LDS at lds_off, packed row-major.
// D# layout per CDNA5 ISA.
// 6-arg builtin form (amdgpu-toolchain / clang-23): groups 0..3 + extra + cpol.
static __device__ inline void tdm_load_tile_2d(
    const __bf16* gbase, unsigned lds_off,
    unsigned tile_k, unsigned tile_rows,
    unsigned tensor_k, unsigned row_stride)
{
  unsigned long long ga = (unsigned long long)(uintptr_t)gbase;
  v4u g0;
  g0[0] = 1u;                                               // count=1, user D#
  g0[1] = lds_off;                                          // lds_addr (bytes)
  g0[2] = (unsigned)ga;                                     // global_addr lo
  g0[3] = (unsigned)((ga >> 32) & 0x1FFFFFFu) | (2u << 30); // addr hi | type=2
  v8i g1;
  g1[0] = (int)(1u << 16);                                  // data_size=1 (2B)
  g1[1] = (int)((tensor_k & 0xFFFFu) << 16);                // tensor_dim0 lo16
  g1[2] = (int)((tensor_k >> 16) | ((tile_rows & 0xFFFFu) << 16)); // d0 hi | tensor_dim1 lo
  g1[3] = (int)((tile_rows >> 16) | (tile_k << 16));        // tensor_dim1 hi | tile_dim0
  g1[4] = (int)(tile_rows & 0xFFFFu);                       // tile_dim1 | tile_dim2=0
  g1[5] = (int)row_stride;                                  // tensor_dim0_stride lo32
  g1[6] = 0;                                                // stride hi | dim1_stride lo
  g1[7] = 0;
  v4i z4 = {0, 0, 0, 0};
  v8i z8 = {0, 0, 0, 0, 0, 0, 0, 0};
  __builtin_amdgcn_tensor_load_to_lds(g0, g1, z4, z4, z8, 0);
}
#endif

// ---------------------------------------------------------------------------
// bf16 GEMM: C[M][N] (f32) = A[M][K] (bf16 row major) @ Bt[N][K]^T
// Block: 256 threads = 8 waves, 128x64 tile. Each wave: 32(M) x 32(N) = 4 WMMA.
// Staging: TDM tensor_load_to_lds when available (wave 0 issues, all compute),
// else per-thread b128 loads + ds_store.
// ---------------------------------------------------------------------------
__global__ __launch_bounds__(256) void pedus_gemm_bf16(
    const __bf16* __restrict__ A, const __bf16* __restrict__ Bt,
    float* __restrict__ C, int Ndim, int Kdim)
{
  __shared__ alignas(16) __bf16 sa[128][32];
  __shared__ alignas(16) __bf16 sb[64][32];
  const int n0   = blockIdx.x * 64;
  const int m0   = blockIdx.y * 128;
  const int tid  = threadIdx.x;
  const int lane = tid & 31;
  const int wid  = tid >> 5;   // 0..7
  const int wm   = wid >> 1;   // 0..3 -> 32 rows each
  const int wn   = wid & 1;    // 0..1 -> 32 cols each
  const int lm   = lane & 15;
  const int lh   = lane >> 4;

  v8f acc00 = {}, acc01 = {}, acc10 = {}, acc11 = {};

#if PEDUS_HAVE_TDM
  const unsigned sa_off = (unsigned)(uintptr_t)(void*)&sa[0][0];
  const unsigned sb_off = (unsigned)(uintptr_t)(void*)&sb[0][0];
#else
  const int r0 = tid >> 2, ch0 = (tid & 3) * 8;             // A chunk 0 (rows 0..63)
  const int r1 = (tid + 256) >> 2, ch1 = ch0;               // A chunk 1 (rows 64..127)
  const int rb = tid >> 2, chb = ch0;                       // B chunk  (rows 0..63)
  const __bf16* gA0 = A  + (size_t)(m0 + r0) * Kdim + ch0;
  const __bf16* gA1 = A  + (size_t)(m0 + r1) * Kdim + ch1;
  const __bf16* gB  = Bt + (size_t)(n0 + rb) * Kdim + chb;
#endif

  for (int k0 = 0; k0 < Kdim; k0 += 32) {
    __syncthreads();
#if PEDUS_HAVE_TDM
    if (wid == 0) {
      tdm_load_tile_2d(A  + (size_t)m0 * Kdim + k0, sa_off, 32u, 128u,
                       (unsigned)Kdim, (unsigned)Kdim);
      tdm_load_tile_2d(Bt + (size_t)n0 * Kdim + k0, sb_off, 32u, 64u,
                       (unsigned)Kdim, (unsigned)Kdim);
      __builtin_amdgcn_s_wait_tensorcnt(0);
    }
#else
    *(B16x8*)&sa[r0][ch0] = *(const B16x8*)(gA0 + k0);
    *(B16x8*)&sa[r1 & 127][ch1] = *(const B16x8*)(gA1 + k0);
    *(B16x8*)&sb[rb][chb] = *(const B16x8*)(gB + k0);
    if (k0 + 32 < Kdim) {
      __builtin_prefetch(gA0 + k0 + 32, 0, 1);
      __builtin_prefetch(gB + k0 + 32, 0, 1);
    }
#endif
    __syncthreads();
    v16bf a0 = load_a_frag(&sa[wm * 32][0],      32, lm, lane);
    v16bf a1 = load_a_frag(&sa[wm * 32 + 16][0], 32, lm, lane);
    v16bf b0 = load_b_frag(&sb[wn * 32][0],      32, lm, lane);
    v16bf b1 = load_b_frag(&sb[wn * 32 + 16][0], 32, lm, lane);
    acc00 = WMMA_BF16(a0, b0, acc00);
    acc01 = WMMA_BF16(a0, b1, acc01);
    acc10 = WMMA_BF16(a1, b0, acc10);
    acc11 = WMMA_BF16(a1, b1, acc11);
  }
  float* Cr0 = C + (size_t)(m0 + wm * 32 + 8 * lh) * Ndim + n0 + wn * 32 + lm;
  float* Cr1 = Cr0 + (size_t)16 * Ndim;
  for (int r = 0; r < 8; ++r) {
    Cr0[(size_t)r * Ndim]      = acc00[r];
    Cr0[(size_t)r * Ndim + 16] = acc01[r];
    Cr1[(size_t)r * Ndim]      = acc10[r];
    Cr1[(size_t)r * Ndim + 16] = acc11[r];
  }
}

// ---------------------------------------------------------------------------
// Flash attention: per (qtile of 64, b*h). 128 threads = 4 waves, 16 q-rows/wave.
// Streams K/V tiles of 32 keys through LDS (with f32->bf16 convert in flight);
// online softmax; QK^T and P@V on WMMA.
// qkv layout: [B*T][2304] f32 with q|k|v blocks of 768, head h at h*64.
// ---------------------------------------------------------------------------
__global__ __launch_bounds__(128) void pedus_flash_attn(
    const float* __restrict__ qkv, float* __restrict__ y)
{
  __shared__ alignas(16) __bf16 sq[64][64];
  __shared__ alignas(16) __bf16 sk[32][64];
  __shared__ alignas(16) __bf16 sv[64][32];   // V transposed: [hd][key]
  __shared__ alignas(16) __bf16 sp[4][16][32];

  const int q0blk = blockIdx.x * 64;
  const int bh = blockIdx.y;
  const int b  = bh / Hm;
  const int h  = bh % Hm;
  const int tid  = threadIdx.x;
  const int lane = tid & 31;
  const int w    = tid >> 5;
  const int lm   = lane & 15;
  const int lh   = lane >> 4;
  const size_t rowbase = (size_t)b * Tm;

  for (int i = 0; i < 32; ++i) {               // stage Q 64x64 -> bf16
    int e = tid * 32 + i;
    int qi = e >> 6, d = e & 63;
    sq[qi][d] = (__bf16)qkv[(rowbase + q0blk + qi) * 2304 + h * 64 + d];
  }
  __syncthreads();

  v16bf qa0 = load_a_frag(&sq[w * 16][0], 64, lm, lane);
  v16bf qa1 = load_a_frag(&sq[w * 16][32], 64, lm, lane);

  float mrun[8], lrun[8];
  v8f o0 = {}, o1 = {}, o2 = {}, o3 = {};
  for (int r = 0; r < 8; ++r) { mrun[r] = -INFINITY; lrun[r] = 0.f; }

  const int q0  = q0blk + w * 16;
  const int nkb = (q0blk + 64) >> 5;
  for (int kbi = 0; kbi < nkb; ++kbi) {
    const int kb = kbi * 32;
    __syncthreads();
    for (int i = 0; i < 16; ++i) {             // stage K 32x64 (natural)
      int e = tid * 16 + i;
      int kr = e >> 6, d = e & 63;
      sk[kr][d] = (__bf16)qkv[(rowbase + kb + kr) * 2304 + 768 + h * 64 + d];
    }
    for (int i = 0; i < 16; ++i) {             // stage V transposed 64x32
      int e = tid * 16 + i;
      int d = e >> 5, kr = e & 31;
      sv[d][kr] = (__bf16)qkv[(rowbase + kb + kr) * 2304 + 1536 + h * 64 + d];
    }
    __syncthreads();

    v8f s0 = {}, s1 = {};
    {
      v16bf k00 = load_b_frag(&sk[0][0],   64, lm, lane);
      v16bf k01 = load_b_frag(&sk[0][32],  64, lm, lane);
      v16bf k10 = load_b_frag(&sk[16][0],  64, lm, lane);
      v16bf k11 = load_b_frag(&sk[16][32], 64, lm, lane);
      s0 = WMMA_BF16(qa0, k00, s0); s0 = WMMA_BF16(qa1, k01, s0);
      s1 = WMMA_BF16(qa0, k10, s1); s1 = WMMA_BF16(qa1, k11, s1);
    }

    __bf16* pw = &sp[w][0][0];
    for (int r = 0; r < 8; ++r) {              // online softmax per row
      int qg = q0 + r + 8 * lh;
      float v0 = s0[r] * 0.125f;               // 1/sqrt(64)
      float v1 = s1[r] * 0.125f;
      if (kb + lm > qg)      v0 = -INFINITY;   // causal mask
      if (kb + 16 + lm > qg) v1 = -INFINITY;
      float rm = fmaxf(v0, v1);
      for (int msk = 8; msk >= 1; msk >>= 1) rm = fmaxf(rm, __shfl_xor(rm, msk, 16));
      float mnew  = fmaxf(mrun[r], rm);
      float alpha = (mrun[r] == -INFINITY) ? 0.f : expf(mrun[r] - mnew);
      float p0 = expf(v0 - mnew);
      float p1 = expf(v1 - mnew);
      float rs = p0 + p1;
      for (int msk = 8; msk >= 1; msk >>= 1) rs += __shfl_xor(rs, msk, 16);
      lrun[r] = lrun[r] * alpha + rs;
      mrun[r] = mnew;
      o0[r] *= alpha; o1[r] *= alpha; o2[r] *= alpha; o3[r] *= alpha;
      pw[(r + 8 * lh) * 32 + lm]      = (__bf16)p0;
      pw[(r + 8 * lh) * 32 + 16 + lm] = (__bf16)p1;
    }
    // wave-local LDS RAW: order P stores before fragment reload
    asm volatile("s_wait_dscnt 0" ::: "memory");
    v16bf pa  = load_a_frag(pw, 32, lm, lane);
    v16bf vb0 = load_b_frag(&sv[0][0],  32, lm, lane);
    v16bf vb1 = load_b_frag(&sv[16][0], 32, lm, lane);
    v16bf vb2 = load_b_frag(&sv[32][0], 32, lm, lane);
    v16bf vb3 = load_b_frag(&sv[48][0], 32, lm, lane);
    o0 = WMMA_BF16(pa, vb0, o0);
    o1 = WMMA_BF16(pa, vb1, o1);
    o2 = WMMA_BF16(pa, vb2, o2);
    o3 = WMMA_BF16(pa, vb3, o3);
  }
  for (int r = 0; r < 8; ++r) {
    float inv = 1.f / lrun[r];
    size_t orow = (rowbase + q0 + r + 8 * lh) * (size_t)Dm + h * 64 + lm;
    y[orow]      = o0[r] * inv;
    y[orow + 16] = o1[r] * inv;
    y[orow + 32] = o2[r] * inv;
    y[orow + 48] = o3[r] * inv;
  }
}

// ---------------------------------------------------------------------------
// LayerNorm over D=768 -> bf16 output (GEMM-ready A matrix)
// ---------------------------------------------------------------------------
__global__ __launch_bounds__(256) void pedus_ln_bf16(
    const float* __restrict__ x, const float* __restrict__ w,
    const float* __restrict__ bias, __bf16* __restrict__ out)
{
  __shared__ float red[256];
  const int row = blockIdx.x, t = threadIdx.x;
  const float* xr = x + (size_t)row * Dm;
  float a0 = xr[t], a1 = xr[t + 256], a2 = xr[t + 512];
  red[t] = a0 + a1 + a2; __syncthreads();
  for (int o = 128; o > 0; o >>= 1) { if (t < o) red[t] += red[t + o]; __syncthreads(); }
  float mean = red[0] * (1.f / 768.f);
  __syncthreads();
  float d0 = a0 - mean, d1 = a1 - mean, d2 = a2 - mean;
  red[t] = d0 * d0 + d1 * d1 + d2 * d2; __syncthreads();
  for (int o = 128; o > 0; o >>= 1) { if (t < o) red[t] += red[t + o]; __syncthreads(); }
  float rstd = rsqrtf(red[0] * (1.f / 768.f) + 1e-5f);
  __bf16* orow = out + (size_t)row * Dm;
  orow[t]       = (__bf16)(d0 * rstd * w[t]       + bias[t]);
  orow[t + 256] = (__bf16)(d1 * rstd * w[t + 256] + bias[t + 256]);
  orow[t + 512] = (__bf16)(d2 * rstd * w[t + 512] + bias[t + 512]);
}

__global__ __launch_bounds__(256) void pedus_embed(
    const int* __restrict__ idx, const float* __restrict__ tok,
    const float* __restrict__ pos, float* __restrict__ x)
{
  const int row = blockIdx.x;
  const int tokid = idx[row];
  const int t = row & (Tm - 1);
  const float* te = tok + (size_t)tokid * Dm;
  const float* pe = pos + (size_t)t * Dm;
  float* xr = x + (size_t)row * Dm;
  for (int d = threadIdx.x; d < Dm; d += 256) xr[d] = te[d] + pe[d];
}

__global__ void pedus_add_inplace(float* __restrict__ x, const float* __restrict__ t, int n) {
  int i = blockIdx.x * 256 + threadIdx.x;
  if (i < n) x[i] += t[i];
}

__global__ void pedus_gelu_bf16(const float* __restrict__ in, __bf16* __restrict__ out, int n) {
  int i = blockIdx.x * 256 + threadIdx.x;
  if (i < n) { float v = in[i]; out[i] = (__bf16)(0.5f * v * (1.f + erff(v * 0.70710678118f))); }
}

__global__ void pedus_cvt_bf16(const float* __restrict__ in, __bf16* __restrict__ out, size_t n) {
  size_t i = (size_t)blockIdx.x * 256 + threadIdx.x;
  if (i < n) out[i] = (__bf16)in[i];
}

// f32 [K][N] -> bf16 [N][K]  (weight transpose+convert; blockIdx.z = layer)
__global__ __launch_bounds__(256) void pedus_cvt_w_t(
    const float* __restrict__ in, __bf16* __restrict__ out, int K, int N)
{
  __shared__ float tile[32][33];
  const size_t loff = (size_t)blockIdx.z * K * N;
  in += loff; out += loff;
  const int n0 = blockIdx.x * 32, k0 = blockIdx.y * 32;
  const int tx = threadIdx.x, ty = threadIdx.y;     // 32x8
  for (int i = 0; i < 4; ++i) {
    int k = ty + i * 8;
    tile[k][tx] = in[(size_t)(k0 + k) * N + n0 + tx];
  }
  __syncthreads();
  for (int i = 0; i < 4; ++i) {
    int n = ty + i * 8;
    out[(size_t)(n0 + n) * K + k0 + tx] = (__bf16)tile[tx][n];
  }
}

// ---------------------------------------------------------------------------
extern "C" void kernel_launch(void* const* d_in, const int* in_sizes, int n_in,
                              void* d_out, int out_size, void* d_ws, size_t ws_size,
                              hipStream_t stream)
{
  (void)in_sizes; (void)n_in; (void)out_size; (void)ws_size;
  const int*   idx     = (const int*)  d_in[0];
  const float* tok_emb = (const float*)d_in[1];
  const float* pos_emb = (const float*)d_in[2];
  const float* ln1_w   = (const float*)d_in[3];
  const float* ln1_b   = (const float*)d_in[4];
  const float* qkv_w   = (const float*)d_in[5];
  const float* out_w   = (const float*)d_in[6];
  const float* ln2_w   = (const float*)d_in[7];
  const float* ln2_b   = (const float*)d_in[8];
  const float* ff1_w   = (const float*)d_in[9];
  const float* ff2_w   = (const float*)d_in[10];
  const float* lnf_w   = (const float*)d_in[11];
  const float* lnf_b   = (const float*)d_in[12];
  float* out = (float*)d_out;

  char* ws = (char*)d_ws;
  size_t off = 0;
  auto carve = [&](size_t bytes) -> char* {
    char* p = ws + off;
    off = (off + bytes + 255) & ~(size_t)255;
    return p;
  };
  __bf16* wqkv_t = (__bf16*)carve((size_t)Lm * 2304 * 768 * 2);
  __bf16* wout_t = (__bf16*)carve((size_t)Lm * 768 * 768 * 2);
  __bf16* wff1_t = (__bf16*)carve((size_t)Lm * 3072 * 768 * 2);
  __bf16* wff2_t = (__bf16*)carve((size_t)Lm * 768 * 3072 * 2);
  __bf16* emb_b  = (__bf16*)carve((size_t)Vm * 768 * 2);     // [V][D] == Bt for lm_head
  float*  x      = (float*) carve((size_t)Mrows * 768 * 4);
  __bf16* hb     = (__bf16*)carve((size_t)Mrows * 768 * 2);
  __bf16* gb     = (__bf16*)carve((size_t)Mrows * 3072 * 2);
  float*  qkv    = (float*) carve((size_t)Mrows * 2304 * 4);
  float*  yatt   = (float*) carve((size_t)Mrows * 768 * 4);
  __bf16* yb     = (__bf16*)carve((size_t)Mrows * 768 * 2);
  float*  t0     = (float*) carve((size_t)Mrows * 768 * 4);
  float*  t1     = (float*) carve((size_t)Mrows * 3072 * 4);

  const dim3 blkT(32, 8);
  pedus_cvt_w_t<<<dim3(2304 / 32, 768 / 32, Lm), blkT, 0, stream>>>(qkv_w, wqkv_t, 768, 2304);
  pedus_cvt_w_t<<<dim3(768 / 32, 768 / 32, Lm),  blkT, 0, stream>>>(out_w, wout_t, 768, 768);
  pedus_cvt_w_t<<<dim3(3072 / 32, 768 / 32, Lm), blkT, 0, stream>>>(ff1_w, wff1_t, 768, 3072);
  pedus_cvt_w_t<<<dim3(768 / 32, 3072 / 32, Lm), blkT, 0, stream>>>(ff2_w, wff2_t, 3072, 768);
  {
    size_t n = (size_t)Vm * 768;
    pedus_cvt_bf16<<<(unsigned)((n + 255) / 256), 256, 0, stream>>>(tok_emb, emb_b, n);
  }
  pedus_embed<<<Mrows, 256, 0, stream>>>(idx, tok_emb, pos_emb, x);

  const int nD = Mrows * 768;
  const int nF = Mrows * 3072;
  for (int l = 0; l < Lm; ++l) {
    pedus_ln_bf16<<<Mrows, 256, 0, stream>>>(x, ln1_w + l * Dm, ln1_b + l * Dm, hb);
    pedus_gemm_bf16<<<dim3(2304 / 64, Mrows / 128), 256, 0, stream>>>(
        hb, wqkv_t + (size_t)l * 2304 * 768, qkv, 2304, 768);
    pedus_flash_attn<<<dim3(Tm / 64, Bmb * Hm), 128, 0, stream>>>(qkv, yatt);
    pedus_cvt_bf16<<<(nD + 255) / 256, 256, 0, stream>>>(yatt, yb, (size_t)nD);
    pedus_gemm_bf16<<<dim3(768 / 64, Mrows / 128), 256, 0, stream>>>(
        yb, wout_t + (size_t)l * 768 * 768, t0, 768, 768);
    pedus_add_inplace<<<(nD + 255) / 256, 256, 0, stream>>>(x, t0, nD);
    pedus_ln_bf16<<<Mrows, 256, 0, stream>>>(x, ln2_w + l * Dm, ln2_b + l * Dm, hb);
    pedus_gemm_bf16<<<dim3(3072 / 64, Mrows / 128), 256, 0, stream>>>(
        hb, wff1_t + (size_t)l * 3072 * 768, t1, 3072, 768);
    pedus_gelu_bf16<<<(nF + 255) / 256, 256, 0, stream>>>(t1, gb, nF);
    pedus_gemm_bf16<<<dim3(768 / 64, Mrows / 128), 256, 0, stream>>>(
        gb, wff2_t + (size_t)l * 768 * 3072, t0, 768, 3072);
    pedus_add_inplace<<<(nD + 255) / 256, 256, 0, stream>>>(x, t0, nD);
  }
  pedus_ln_bf16<<<Mrows, 256, 0, stream>>>(x, lnf_w, lnf_b, hb);
  pedus_gemm_bf16<<<dim3(Vm / 64, Mrows / 128), 256, 0, stream>>>(
      hb, emb_b, out, Vm, 768);
}
